// Encoder_13219909337540
// MI455X (gfx1250) — matrix operand order repaired and checked by reading, not compiled, not run on hardware
//
#include <hip/hip_runtime.h>

#define N_NODES 100000
#define N_EDGES 1600000
#define DIM 128

typedef __attribute__((ext_vector_type(16))) _Float16 v16h;
typedef __attribute__((ext_vector_type(8)))  _Float16 v8h;
typedef __attribute__((ext_vector_type(8)))  float    v8f;

// ---------------------------------------------------------------- utilities
__global__ void zero_f32(float* __restrict__ p, long n) {
  long i = (long)blockIdx.x * blockDim.x + threadIdx.x;
  if (i < n) p[i] = 0.0f;
}

// one thread per edge: deg[dst] += 1
__global__ void deg_count(const int* __restrict__ dst, float* __restrict__ deg, int E) {
  int e = blockIdx.x * blockDim.x + threadIdx.x;
  if (e < E) atomicAdd(&deg[dst[e]], 1.0f);
}

// invdeg[n] = 1 / max(deg[n], 1)
__global__ void finalize_invdeg(float* __restrict__ deg, int N) {
  int n = blockIdx.x * blockDim.x + threadIdx.x;
  if (n < N) deg[n] = 1.0f / fmaxf(deg[n], 1.0f);
}

// transpose + convert 128x128 f32 weight -> f16 WT[c][k] = W[k][c]
__global__ void cvt_weightT(const float* __restrict__ W, _Float16* __restrict__ WT) {
  int t = blockIdx.x * blockDim.x + threadIdx.x;   // 0 .. 16383
  int k = t & (DIM - 1);
  int c = t >> 7;
  WT[(size_t)c * DIM + k] = (_Float16)W[(size_t)k * DIM + c];
}

// ------------------------------------------------------- edge scatter (mean)
// one wave (32 lanes) per edge; lane handles 4 consecutive features.
// x table (51MB) and agg (51MB) both fit in the 192MB L2, so the gather and
// the f32 atomics stay L2-resident on MI455X.
__global__ void scatter_edges(const float* __restrict__ h,
                              const int* __restrict__ src,
                              const int* __restrict__ dst,
                              const float* __restrict__ ew,
                              float* __restrict__ agg, int E) {
  int gid  = blockIdx.x * blockDim.x + threadIdx.x;
  int e    = gid >> 5;
  int lane = gid & 31;
  if (e >= E) return;
  int   s = src[e];
  int   d = dst[e];
  float w = ew[e];
  float4 v = *(const float4*)(h + (size_t)s * DIM + lane * 4);
  float* a = agg + (size_t)d * DIM + lane * 4;
  atomicAdd(a + 0, v.x * w);
  atomicAdd(a + 1, v.y * w);
  atomicAdd(a + 2, v.z * w);
  atomicAdd(a + 3, v.w * w);
}

// convert agg*invdeg and h_in to f16 row-major for WMMA A operands
__global__ void cvt_inputs(const float* __restrict__ agg,
                           const float* __restrict__ invdeg,
                           const float* __restrict__ h,
                           _Float16* __restrict__ aggA,
                           _Float16* __restrict__ hA, int N) {
  long t = (long)blockIdx.x * blockDim.x + threadIdx.x;
  long n = t >> 5;
  int  q = (int)(t & 31);
  if (n >= N) return;
  float  inv = invdeg[n];
  size_t off = (size_t)n * DIM + q * 4;
  float4 a  = *(const float4*)(agg + off);
  float4 xx = *(const float4*)(h + off);
  _Float16* pa = aggA + off;
  _Float16* ph = hA + off;
  pa[0] = (_Float16)(a.x * inv); pa[1] = (_Float16)(a.y * inv);
  pa[2] = (_Float16)(a.z * inv); pa[3] = (_Float16)(a.w * inv);
  ph[0] = (_Float16)xx.x; ph[1] = (_Float16)xx.y;
  ph[2] = (_Float16)xx.z; ph[3] = (_Float16)xx.w;
}

// ------------------------------------------------------------- fused GEMMs
// out[16 rows x 128 cols per wave] = act( aggA @ WlT^T + bias + hA @ WrT^T )
// WT is stored [n][k] so the B fragment (32x16 f16, lane L = column L%16,
// K-range (L/16)*16) is one contiguous 32B load. A fragment (16x32 f16,
// lane L = row L%16, K offsets (L/16)*16 and +8) is two contiguous 16B loads.
__global__ void sage_gemm(const _Float16* __restrict__ aggA,
                          const _Float16* __restrict__ hA,
                          const _Float16* __restrict__ WlT,
                          const _Float16* __restrict__ WrT,
                          const float* __restrict__ bias,
                          float* __restrict__ out,
                          int do_relu, int nrows) {
  const int wave = (int)((blockIdx.x * blockDim.x + threadIdx.x) >> 5);
  const int lane = threadIdx.x & 31;
  const int row0 = wave * 16;
  if (row0 >= nrows) return;           // wave-uniform: EXEC stays all-ones
  const int hs  = lane >> 4;           // half-wave select (0/1)
  const int l16 = lane & 15;

  v8f acc[8];
#pragma unroll
  for (int i = 0; i < 8; ++i) acc[i] = (v8f){0.f,0.f,0.f,0.f,0.f,0.f,0.f,0.f};

  const size_t arow = (size_t)(row0 + l16) * DIM;

#pragma unroll
  for (int path = 0; path < 2; ++path) {
    const _Float16* A = path ? hA : aggA;
    const _Float16* W = path ? WrT : WlT;
#pragma unroll
    for (int kk = 0; kk < DIM; kk += 32) {
      const _Float16* ap = A + arow + kk + hs * 16;
      v8h alo = *(const v8h*)(ap);
      v8h ahi = *(const v8h*)(ap + 8);
      v16h afrag;
#pragma unroll
      for (int i = 0; i < 8; ++i) { afrag[i] = alo[i]; afrag[8 + i] = ahi[i]; }
#pragma unroll
      for (int ct = 0; ct < 8; ++ct) {
        v16h bfrag = *(const v16h*)(W + (size_t)(ct * 16 + l16) * DIM + kk + hs * 16);
        acc[ct] = __builtin_amdgcn_wmma_f32_16x16x32_f16(
            false, afrag, false, bfrag, (short)0, acc[ct], false, false);
      }
    }
  }

  // C/D layout: lane L, vgpr r -> (M = r + (L/16)*8, N = L%16)
#pragma unroll
  for (int ct = 0; ct < 8; ++ct) {
    const int   col = ct * 16 + l16;
    const float b   = bias[col];
#pragma unroll
    for (int r = 0; r < 8; ++r) {
      float v = acc[ct][r] + b;
      if (do_relu) v = fmaxf(v, 0.0f);
      out[(size_t)(row0 + hs * 8 + r) * DIM + col] = v;
    }
  }
}

// ------------------------------------------------------------------- driver
extern "C" void kernel_launch(void* const* d_in, const int* in_sizes, int n_in,
                              void* d_out, int out_size, void* d_ws, size_t ws_size,
                              hipStream_t stream) {
  const float* x  = (const float*)d_in[0];
  const int*   ei = (const int*)d_in[1];        // [2][E] (int indices)
  const float* ew = (const float*)d_in[2];
  const float* Wl[3] = {(const float*)d_in[3], (const float*)d_in[6], (const float*)d_in[9]};
  const float* bl[3] = {(const float*)d_in[4], (const float*)d_in[7], (const float*)d_in[10]};
  const float* Wr[3] = {(const float*)d_in[5], (const float*)d_in[8], (const float*)d_in[11]};
  float* out = (float*)d_out;

  const int* src = ei;
  const int* dst = ei + N_EDGES;

  // workspace carve-out (256B aligned regions), ~154 MB total
  char*  ws  = (char*)d_ws;
  size_t off = 0;
  auto take = [&](size_t bytes) -> char* {
    char* p = ws + off;
    off = (off + bytes + 255) & ~(size_t)255;
    return p;
  };
  float*    agg    = (float*)take((size_t)N_NODES * DIM * sizeof(float));
  float*    h1     = (float*)take((size_t)N_NODES * DIM * sizeof(float));
  float*    invdeg = (float*)take((size_t)N_NODES * sizeof(float));
  _Float16* aggA   = (_Float16*)take((size_t)N_NODES * DIM * sizeof(_Float16));
  _Float16* hA     = (_Float16*)take((size_t)N_NODES * DIM * sizeof(_Float16));
  _Float16* WlT[3];
  _Float16* WrT[3];
  for (int i = 0; i < 3; ++i) {
    WlT[i] = (_Float16*)take((size_t)DIM * DIM * sizeof(_Float16));
    WrT[i] = (_Float16*)take((size_t)DIM * DIM * sizeof(_Float16));
  }
  (void)in_sizes; (void)n_in; (void)out_size; (void)ws_size;

  const dim3 B(256);
  const long nd = (long)N_NODES * DIM;

  // weights -> transposed f16 (once per call; deterministic)
  for (int i = 0; i < 3; ++i) {
    cvt_weightT<<<(DIM * DIM + 255) / 256, B, 0, stream>>>(Wl[i], WlT[i]);
    cvt_weightT<<<(DIM * DIM + 255) / 256, B, 0, stream>>>(Wr[i], WrT[i]);
  }

  // degrees -> invdeg (stored in place)
  zero_f32<<<(N_NODES + 255) / 256, B, 0, stream>>>(invdeg, N_NODES);
  deg_count<<<(N_EDGES + 255) / 256, B, 0, stream>>>(dst, invdeg, N_EDGES);
  finalize_invdeg<<<(N_NODES + 255) / 256, B, 0, stream>>>(invdeg, N_NODES);

  const float* hin = x;
  float* houts[3] = {h1, out, out};   // layer-2 output aliases d_out; safe:
                                      // layer-3 reads it only before its GEMM writes
  for (int L = 0; L < 3; ++L) {
    zero_f32<<<(int)((nd + 255) / 256), B, 0, stream>>>(agg, nd);
    scatter_edges<<<(N_EDGES * 32 + 255) / 256, B, 0, stream>>>(hin, src, dst, ew, agg, N_EDGES);
    cvt_inputs<<<(N_NODES * 32 + 255) / 256, B, 0, stream>>>(agg, invdeg, hin, aggA, hA, N_NODES);
    sage_gemm<<<((N_NODES / 16) + 7) / 8, B, 0, stream>>>(
        aggA, hA, WlT[L], WrT[L], bl[L], houts[L], (L < 2) ? 1 : 0, N_NODES);
    hin = houts[L];
  }
}
